// MultiHeadAttention_472446403096
// MI455X (gfx1250) — matrix operand order repaired and checked by reading, not compiled, not run on hardware
//
#include <hip/hip_runtime.h>
#include <math.h>

// Problem constants (from reference)
#define BATCH   2
#define SEQ     2048
#define DMODEL  1024
#define NHEAD   16
#define DK      64
#define NQKV    3072

typedef __bf16 bf16_t;
typedef __attribute__((ext_vector_type(16))) __bf16 v16bf;
typedef __attribute__((ext_vector_type(8)))  float  v8f;

struct FragBits { uint4 lo, hi; };

#define WMMA_BF16(A, B, C) \
  __builtin_amdgcn_wmma_f32_16x16x32_bf16(false, (A), false, (B), (short)0, (C), false, false)

// ---- async global->LDS (gfx1250) with safe fallback --------------------
#define AS1 __attribute__((address_space(1)))
#define AS3 __attribute__((address_space(3)))

typedef int v4i_ __attribute__((__vector_size__(16)));  // matches builtin param type

#if defined(__has_builtin)
#if __has_builtin(__builtin_amdgcn_global_load_async_to_lds_b128)
#define HAVE_ASYNC_LDS 1
#endif
#endif

__device__ __forceinline__ void copy16_g2l(const void* g, void* l) {
#ifdef HAVE_ASYNC_LDS
  __builtin_amdgcn_global_load_async_to_lds_b128(
      (AS1 v4i_*)(unsigned long long)(size_t)g,   // global AS pointer via integer round-trip
      (AS3 v4i_*)(unsigned)(size_t)l,             // LDS address = low 32 bits of generic ptr
      0, 0);
#else
  *(uint4*)l = *(const uint4*)g;
#endif
}

__device__ __forceinline__ void wait_async_all() {
#ifdef HAVE_ASYNC_LDS
#if __has_builtin(__builtin_amdgcn_s_wait_asynccnt)
  __builtin_amdgcn_s_wait_asynccnt(0);
#else
  asm volatile("s_wait_asynccnt 0x0" ::: "memory");
#endif
#endif
}

// ------------------------------------------------------------------------
__device__ __forceinline__ unsigned short f32_to_bf16_bits(float f) {
  unsigned int u = __builtin_bit_cast(unsigned int, f);
  u += 0x7FFFu + ((u >> 16) & 1u);   // round-to-nearest-even
  return (unsigned short)(u >> 16);
}

__device__ __forceinline__ unsigned int pack_bf16x2(float a, float b) {
  return (unsigned int)f32_to_bf16_bits(a) | ((unsigned int)f32_to_bf16_bits(b) << 16);
}

// A fragment: 16x32 bf16, M = lane&15, K-halves selected by lane>>4.
__device__ __forceinline__ v16bf load_frag_a(const unsigned short* tile, int strideHalf,
                                             int lane, int kByte) {
  const char* base = (const char*)tile + (lane & 15) * (strideHalf * 2) + kByte + ((lane >> 4) * 16);
  FragBits f;
  f.lo = *(const uint4*)(base);
  f.hi = *(const uint4*)(base + 32);
  return __builtin_bit_cast(v16bf, f);
}

// B fragment from a transposed tile Bt[n][k]: N = lane&15, K-half by lane>>4.
__device__ __forceinline__ v16bf load_frag_bt(const unsigned short* tileT, int strideHalf,
                                              int lane, int kByte) {
  const char* base = (const char*)tileT + (lane & 15) * (strideHalf * 2) + kByte + ((lane >> 4) * 16);
  FragBits f;
  f.lo = *(const uint4*)(base);
  f.hi = *(const uint4*)(base + 32);
  return __builtin_bit_cast(v16bf, f);
}

// ---------------------------------------------------------------- convert (vectorized)
__global__ __launch_bounds__(256) void cvt_f32_bf16_kernel(const float4* __restrict__ src,
                                                           uint2* __restrict__ dst, int n4) {
  int i = blockIdx.x * blockDim.x + threadIdx.x;
  int stride = gridDim.x * blockDim.x;
  for (; i < n4; i += stride) {
    float4 v = src[i];
    uint2 o;
    o.x = pack_bf16x2(v.x, v.y);
    o.y = pack_bf16x2(v.z, v.w);
    dst[i] = o;
  }
}

// ---------------------------------------------------------------- QKV GEMM 128x128
// C[4096 x 3072] = xb @ Wab (+bias), scatter to q/k/v [B,H,S,DK] bf16.
__global__ __launch_bounds__(256) void qkv_gemm_kernel(const unsigned short* __restrict__ xb,
                                                       const unsigned short* __restrict__ Wab,
                                                       const float* __restrict__ b_attn,
                                                       unsigned short* __restrict__ qb,
                                                       unsigned short* __restrict__ kb,
                                                       unsigned short* __restrict__ vb) {
  __shared__ __align__(16) unsigned short smA[128 * 40];   // 128x32 bf16, row stride 80B
  __shared__ __align__(16) unsigned short smB[128 * 40];   // Bt[n=128][k=32]
  const int n0 = blockIdx.x * 128;
  const int i0 = blockIdx.y * 128;
  const int t = threadIdx.x, lane = t & 31, wave = t >> 5;
  const int wm = (wave >> 1) * 32, wn = (wave & 1) * 64;   // wave tile 32x64
  v8f acc[2][4] = {};

  for (int k0 = 0; k0 < DMODEL; k0 += 32) {
    // A tile: 128x32, direct global->LDS (async on gfx1250)
#pragma unroll
    for (int i = 0; i < 2; ++i) {
      const int idx = t + i * 256;
      const int row = idx >> 2, c = (idx & 3) * 8;
      copy16_g2l(xb + (size_t)(i0 + row) * DMODEL + k0 + c, smA + row * 40 + c);
    }
    // B tile: 32x128, transpose into LDS
#pragma unroll
    for (int i = 0; i < 2; ++i) {
      const int idx = t + i * 256;
      const int k = idx >> 4, c0 = (idx & 15) * 8;
      uint4 bv = *(const uint4*)(Wab + (size_t)(k0 + k) * NQKV + n0 + c0);
      const unsigned short* bs = (const unsigned short*)&bv;
#pragma unroll
      for (int j = 0; j < 8; ++j) smB[(c0 + j) * 40 + k] = bs[j];
      if (k0 + 32 < DMODEL)
        __builtin_prefetch(Wab + (size_t)(k0 + 32 + k) * NQKV + n0 + c0, 0, 3);
    }
    wait_async_all();
    __syncthreads();

    v16bf af[2], bf[4];
#pragma unroll
    for (int tm = 0; tm < 2; ++tm) af[tm] = load_frag_a(smA + (wm + tm * 16) * 40, 40, lane, 0);
#pragma unroll
    for (int tn = 0; tn < 4; ++tn) bf[tn] = load_frag_bt(smB + (wn + tn * 16) * 40, 40, lane, 0);
#pragma unroll
    for (int tm = 0; tm < 2; ++tm)
#pragma unroll
      for (int tn = 0; tn < 4; ++tn)
        acc[tm][tn] = WMMA_BF16(af[tm], bf[tn], acc[tm][tn]);
    __syncthreads();
  }

  const int n = lane & 15, mb = (lane >> 4) * 8;
  const int b = i0 >> 11;                  // 2048 rows per batch; i0 multiple of 128
#pragma unroll
  for (int tm = 0; tm < 2; ++tm) {
    const int s0 = (i0 & (SEQ - 1)) + wm + tm * 16 + mb;
#pragma unroll
    for (int tn = 0; tn < 4; ++tn) {
      const int j = n0 + wn + tn * 16 + n;
      const int which = j >> 10, d = j & 1023, h = d >> 6, dk = d & 63;
      unsigned short* dst = (which == 0) ? qb : ((which == 1) ? kb : vb);
      const float bias = b_attn[j];
      size_t base = (((size_t)(b * NHEAD + h)) * SEQ + s0) * DK + dk;
#pragma unroll
      for (int r = 0; r < 8; ++r)
        dst[base + (size_t)r * DK] = f32_to_bf16_bits(acc[tm][tn][r] + bias);
    }
  }
}

// ---------------------------------------------------------------- scores = q k^T * scale + mask (128x128, K=64)
__global__ __launch_bounds__(256) void scores_gemm_kernel(const unsigned short* __restrict__ qb,
                                                          const unsigned short* __restrict__ kb,
                                                          const float* __restrict__ mask,
                                                          float* __restrict__ scores) {
  __shared__ __align__(16) unsigned short smQ[128 * 72];  // 128 rows x 64 bf16, stride 144B
  __shared__ __align__(16) unsigned short smK[128 * 72];  // k rows ARE the transposed B tile
  const int j0 = blockIdx.x * 128;
  const int i0 = blockIdx.y * 128;
  const int bh = blockIdx.z;
  const int bidx = bh >> 4;
  const size_t headOff = (size_t)bh * SEQ * DK;
  const int t = threadIdx.x, lane = t & 31, wave = t >> 5;
  const int wm = (wave >> 1) * 32, wn = (wave & 1) * 64;

#pragma unroll
  for (int i = 0; i < 4; ++i) {
    const int idx = t + i * 256;
    const int row = idx >> 3, c = (idx & 7) * 8;
    copy16_g2l(qb + headOff + (size_t)(i0 + row) * DK + c, smQ + row * 72 + c);
    copy16_g2l(kb + headOff + (size_t)(j0 + row) * DK + c, smK + row * 72 + c);
  }
  wait_async_all();
  __syncthreads();

  v8f acc[2][4] = {};
#pragma unroll
  for (int kc = 0; kc < 2; ++kc) {
    v16bf af[2], bf[4];
#pragma unroll
    for (int tm = 0; tm < 2; ++tm) af[tm] = load_frag_a(smQ + (wm + tm * 16) * 72, 72, lane, kc * 64);
#pragma unroll
    for (int tn = 0; tn < 4; ++tn) bf[tn] = load_frag_bt(smK + (wn + tn * 16) * 72, 72, lane, kc * 64);
#pragma unroll
    for (int tm = 0; tm < 2; ++tm)
#pragma unroll
      for (int tn = 0; tn < 4; ++tn)
        acc[tm][tn] = WMMA_BF16(af[tm], bf[tn], acc[tm][tn]);
  }

  const int n = lane & 15, mb = (lane >> 4) * 8;
  const float scale = 0.125f;  // 1/sqrt(64)
#pragma unroll
  for (int tm = 0; tm < 2; ++tm) {
    const int irow = i0 + wm + tm * 16 + mb;
#pragma unroll
    for (int tn = 0; tn < 4; ++tn) {
      const int j = j0 + wn + tn * 16 + n;
      const float mterm = (1.0f - mask[bidx * SEQ + j]) * -10000.0f;
      size_t base = ((size_t)bh * SEQ + irow) * SEQ + j;
#pragma unroll
      for (int r = 0; r < 8; ++r)
        scores[base + (size_t)r * SEQ] = acc[tm][tn][r] * scale + mterm;
    }
  }
}

// ---------------------------------------------------------------- softmax rows (in place, fp32)
__global__ __launch_bounds__(256) void softmax_rows_kernel(float* __restrict__ p) {
  __shared__ float red[256];
  float* row = p + (size_t)blockIdx.x * SEQ;
  const int t = threadIdx.x;
  float v[8];
  float mx = -3.0e38f;
#pragma unroll
  for (int j = 0; j < 8; ++j) { v[j] = row[t + j * 256]; mx = fmaxf(mx, v[j]); }
  red[t] = mx; __syncthreads();
  for (int s = 128; s > 0; s >>= 1) { if (t < s) red[t] = fmaxf(red[t], red[t + s]); __syncthreads(); }
  mx = red[0]; __syncthreads();
  float sum = 0.0f;
#pragma unroll
  for (int j = 0; j < 8; ++j) { v[j] = __expf(v[j] - mx); sum += v[j]; }
  red[t] = sum; __syncthreads();
  for (int s = 128; s > 0; s >>= 1) { if (t < s) red[t] += red[t + s]; __syncthreads(); }
  const float inv = 1.0f / red[0];
#pragma unroll
  for (int j = 0; j < 8; ++j) row[t + j * 256] = v[j] * inv;
}

// ---------------------------------------------------------------- out_head = P @ v (128x64 tiles)
__global__ __launch_bounds__(256) void pv_gemm_kernel(const float* __restrict__ probs,
                                                      const unsigned short* __restrict__ vb,
                                                      unsigned short* __restrict__ ab) {
  __shared__ __align__(16) unsigned short smP[128 * 40];
  __shared__ __align__(16) unsigned short smV[64 * 40];  // Bt[dk][s]
  const int i0 = blockIdx.x * 128;
  const int bh = blockIdx.y;
  const int b = bh >> 4, h = bh & 15;
  const size_t pBase = (size_t)bh * SEQ * SEQ;
  const size_t vBase = (size_t)bh * SEQ * DK;
  const int t = threadIdx.x, lane = t & 31, wave = t >> 5;
  const int wm = (wave >> 1) * 32, wn = (wave & 1) * 32;  // wave tile 32x32
  v8f acc[2][2] = {};

  for (int k0 = 0; k0 < SEQ; k0 += 32) {
    // P tile: 128x32 fp32 -> bf16 in LDS
#pragma unroll
    for (int i = 0; i < 2; ++i) {
      const int idx = t + i * 256;
      const int row = idx >> 2, c = (idx & 3) * 8;
      const float4* ps = (const float4*)(probs + pBase + (size_t)(i0 + row) * SEQ + k0 + c);
      float4 p0 = ps[0], p1 = ps[1];
      uint4 pk;
      pk.x = pack_bf16x2(p0.x, p0.y); pk.y = pack_bf16x2(p0.z, p0.w);
      pk.z = pack_bf16x2(p1.x, p1.y); pk.w = pack_bf16x2(p1.z, p1.w);
      *(uint4*)(smP + row * 40 + c) = pk;
      if (k0 + 32 < SEQ)
        __builtin_prefetch(probs + pBase + (size_t)(i0 + row) * SEQ + k0 + 32 + c, 0, 3);
    }
    // V tile: 32x64 -> transpose into LDS
    {
      const int k = t >> 3, c0 = (t & 7) * 8;
      uint4 vv = *(const uint4*)(vb + vBase + (size_t)(k0 + k) * DK + c0);
      const unsigned short* vs = (const unsigned short*)&vv;
#pragma unroll
      for (int j = 0; j < 8; ++j) smV[(c0 + j) * 40 + k] = vs[j];
    }
    __syncthreads();

    v16bf af[2], bf[2];
#pragma unroll
    for (int tm = 0; tm < 2; ++tm) af[tm] = load_frag_a(smP + (wm + tm * 16) * 40, 40, lane, 0);
#pragma unroll
    for (int tn = 0; tn < 2; ++tn) bf[tn] = load_frag_bt(smV + (wn + tn * 16) * 40, 40, lane, 0);
#pragma unroll
    for (int tm = 0; tm < 2; ++tm)
#pragma unroll
      for (int tn = 0; tn < 2; ++tn)
        acc[tm][tn] = WMMA_BF16(af[tm], bf[tn], acc[tm][tn]);
    __syncthreads();
  }

  const int n = lane & 15, mb = (lane >> 4) * 8;
#pragma unroll
  for (int tm = 0; tm < 2; ++tm) {
    const int irow = i0 + wm + tm * 16 + mb;
#pragma unroll
    for (int tn = 0; tn < 2; ++tn) {
      const int dk = wn + tn * 16 + n;
      size_t base = ((size_t)b * SEQ + irow) * DMODEL + h * DK + dk;
#pragma unroll
      for (int r = 0; r < 8; ++r)
        ab[base + (size_t)r * DMODEL] = f32_to_bf16_bits(acc[tm][tn][r]);
    }
  }
}

// ---------------------------------------------------------------- final projection 128x128 (fp32 out)
__global__ __launch_bounds__(256) void proj_gemm_kernel(const unsigned short* __restrict__ ab,
                                                        const unsigned short* __restrict__ Wpb,
                                                        const float* __restrict__ b_proj,
                                                        float* __restrict__ out) {
  __shared__ __align__(16) unsigned short smA[128 * 40];
  __shared__ __align__(16) unsigned short smB[128 * 40];
  const int n0 = blockIdx.x * 128;
  const int i0 = blockIdx.y * 128;
  const int t = threadIdx.x, lane = t & 31, wave = t >> 5;
  const int wm = (wave >> 1) * 32, wn = (wave & 1) * 64;
  v8f acc[2][4] = {};

  for (int k0 = 0; k0 < DMODEL; k0 += 32) {
#pragma unroll
    for (int i = 0; i < 2; ++i) {
      const int idx = t + i * 256;
      const int row = idx >> 2, c = (idx & 3) * 8;
      copy16_g2l(ab + (size_t)(i0 + row) * DMODEL + k0 + c, smA + row * 40 + c);
    }
#pragma unroll
    for (int i = 0; i < 2; ++i) {
      const int idx = t + i * 256;
      const int k = idx >> 4, c0 = (idx & 15) * 8;
      uint4 bv = *(const uint4*)(Wpb + (size_t)(k0 + k) * DMODEL + n0 + c0);
      const unsigned short* bs = (const unsigned short*)&bv;
#pragma unroll
      for (int j = 0; j < 8; ++j) smB[(c0 + j) * 40 + k] = bs[j];
      if (k0 + 32 < DMODEL)
        __builtin_prefetch(Wpb + (size_t)(k0 + 32 + k) * DMODEL + n0 + c0, 0, 3);
    }
    wait_async_all();
    __syncthreads();

    v16bf af[2], bf[4];
#pragma unroll
    for (int tm = 0; tm < 2; ++tm) af[tm] = load_frag_a(smA + (wm + tm * 16) * 40, 40, lane, 0);
#pragma unroll
    for (int tn = 0; tn < 4; ++tn) bf[tn] = load_frag_bt(smB + (wn + tn * 16) * 40, 40, lane, 0);
#pragma unroll
    for (int tm = 0; tm < 2; ++tm)
#pragma unroll
      for (int tn = 0; tn < 4; ++tn)
        acc[tm][tn] = WMMA_BF16(af[tm], bf[tn], acc[tm][tn]);
    __syncthreads();
  }

  const int n = lane & 15, mb = (lane >> 4) * 8;
#pragma unroll
  for (int tm = 0; tm < 2; ++tm) {
    const int irow = i0 + wm + tm * 16 + mb;
#pragma unroll
    for (int tn = 0; tn < 4; ++tn) {
      const int j = n0 + wn + tn * 16 + n;
      const float bias = b_proj[j];
      size_t base = (size_t)irow * DMODEL + j;
#pragma unroll
      for (int r = 0; r < 8; ++r)
        out[base + (size_t)r * DMODEL] = acc[tm][tn][r] + bias;
    }
  }
}

// ---------------------------------------------------------------- launch
extern "C" void kernel_launch(void* const* d_in, const int* in_sizes, int n_in,
                              void* d_out, int out_size, void* d_ws, size_t ws_size,
                              hipStream_t stream) {
  const float* x      = (const float*)d_in[0];
  const float* mask   = (const float*)d_in[1];
  const float* W_attn = (const float*)d_in[2];
  const float* b_attn = (const float*)d_in[3];
  const float* W_proj = (const float*)d_in[4];
  const float* b_proj = (const float*)d_in[5];

  float* out    = (float*)d_out;                               // [B,S,D]
  float* scores = out + (size_t)BATCH * SEQ * DMODEL;          // [B,H,S,S]

  // Workspace layout (u16 elements): 48 MB total
  unsigned short* xb  = (unsigned short*)d_ws;                 // [4096,1024]
  unsigned short* Wab = xb  + (size_t)BATCH * SEQ * DMODEL;    // [1024,3072]
  unsigned short* Wpb = Wab + (size_t)DMODEL * NQKV;           // [1024,1024]
  unsigned short* qb  = Wpb + (size_t)DMODEL * DMODEL;         // [B,H,S,DK] x3
  unsigned short* kb  = qb  + (size_t)BATCH * SEQ * DMODEL;
  unsigned short* vb  = kb  + (size_t)BATCH * SEQ * DMODEL;
  unsigned short* ab  = vb  + (size_t)BATCH * SEQ * DMODEL;    // [B*S, D] merged heads

  cvt_f32_bf16_kernel<<<1024, 256, 0, stream>>>((const float4*)x,      (uint2*)xb,
                                                (BATCH * SEQ * DMODEL) / 4);
  cvt_f32_bf16_kernel<<<1024, 256, 0, stream>>>((const float4*)W_attn, (uint2*)Wab,
                                                (DMODEL * NQKV) / 4);
  cvt_f32_bf16_kernel<<<1024, 256, 0, stream>>>((const float4*)W_proj, (uint2*)Wpb,
                                                (DMODEL * DMODEL) / 4);

  qkv_gemm_kernel<<<dim3(NQKV / 128, (BATCH * SEQ) / 128), 256, 0, stream>>>(
      xb, Wab, b_attn, qb, kb, vb);

  scores_gemm_kernel<<<dim3(SEQ / 128, SEQ / 128, BATCH * NHEAD), 256, 0, stream>>>(
      qb, kb, mask, scores);

  softmax_rows_kernel<<<BATCH * NHEAD * SEQ, 256, 0, stream>>>(scores);

  pv_gemm_kernel<<<dim3(SEQ / 128, BATCH * NHEAD), 256, 0, stream>>>(scores, vb, ab);

  proj_gemm_kernel<<<dim3(DMODEL / 128, (BATCH * SEQ) / 128), 256, 0, stream>>>(
      ab, Wpb, b_proj, out);
}
